// CapsGNN_68487548502403
// MI455X (gfx1250) — matrix operand order repaired and checked by reading, not compiled, not run on hardware
//
#include <hip/hip_runtime.h>
#include <hip/hip_bf16.h>

// ---------------------------------------------------------------------------
// GCN forward on MI455X (gfx1250, wave32).
//  - Dense transforms: v_wmma_f32_16x16x32_bf16, fp32 accumulate.
//  - Message passing: fp32 hardware atomics (global_atomic_add_f32),
//    h/agg (25.6 MB each) are L2-resident on the 192 MB L2.
// ---------------------------------------------------------------------------

typedef __attribute__((ext_vector_type(16))) __bf16        v16bf;
typedef __attribute__((ext_vector_type(8)))  float         v8f;
typedef __attribute__((ext_vector_type(8)))  unsigned int  v8u;

#define F      128   // feature width
#define PAIRS  64    // bf16 pairs per row (128/2)

__device__ __forceinline__ unsigned short f32_to_bf16(float f) {
  unsigned int u = __float_as_uint(f);
  u += 0x7FFFu + ((u >> 16) & 1u);       // round-to-nearest-even
  return (unsigned short)(u >> 16);
}
__device__ __forceinline__ unsigned int pack2(float lo, float hi) {
  return (unsigned int)f32_to_bf16(lo) | ((unsigned int)f32_to_bf16(hi) << 16);
}
__device__ __forceinline__ void atomic_add_f32(float* p, float v) {
  unsafeAtomicAdd(p, v);                 // hw global_atomic_add_f32
}

// ---------------- degree / normalization ----------------------------------
__global__ void zero_f32_kernel(float* __restrict__ p, int n) {
  int i = blockIdx.x * blockDim.x + threadIdx.x;
  if (i < n) p[i] = 0.0f;
}

__global__ void deg_kernel(const long long* __restrict__ edges, float* __restrict__ deg, int E) {
  int e = blockIdx.x * blockDim.x + threadIdx.x;
  if (e < E) atomic_add_f32(&deg[(int)edges[(size_t)E + e]], 1.0f);
}

__global__ void dis_kernel(float* __restrict__ d, int n) {
  int i = blockIdx.x * blockDim.x + threadIdx.x;
  if (i < n) d[i] = rsqrtf(d[i] + 1.0f);
}

// ---------------- weight pre-pack into WMMA B-fragment layout --------------
// Layout per layer: [ntile 0..7][kstep 0..3][lane 0..31][j 0..7] dwords.
// B 32x16 packed in this code's contiguous k order: lanes 0-15 hold K=0..15, lanes 16-31 hold K=16..31;
// VGPR j of a lane holds pair {K=2j+16*half, K+1}, column n = 16*t + (lane&15).
__global__ void pack_w_kernel(const float* __restrict__ W, unsigned int* __restrict__ wpack) {
  int gid = blockIdx.x * blockDim.x + threadIdx.x;     // 0 .. 3*8192-1
  int l = gid >> 13;
  int u = gid & 8191;
  int j    = u & 7;
  int lane = (u >> 3) & 31;
  int s    = (u >> 8) & 3;
  int t    = (u >> 10) & 7;
  int n = t * 16 + (lane & 15);
  int k = s * 32 + 16 * (lane >> 4) + 2 * j;
  const float* Wl = W + (size_t)l * F * F;             // [k][n] row-major
  wpack[gid] = pack2(Wl[k * F + n], Wl[(k + 1) * F + n]);
}

// ---------------- h = relu?(x) @ W via WMMA --------------------------------
// One block = 16-row strip; 8 waves cover n-tiles 0..7; K=128 in 4 steps.
__global__ __launch_bounds__(256) void gcn_gemm_bf16(
    const float* __restrict__ x, const unsigned int* __restrict__ wpack,
    float* __restrict__ h, int apply_relu, int nrows)
{
  __shared__ unsigned int ldsx[16 * PAIRS];            // 16 rows x 64 bf16-pairs
  const int tid = threadIdx.x;
  const size_t row0 = (size_t)blockIdx.x * 16;

  { // cooperative stage: 16x128 fp32 -> bf16 pairs in LDS (ReLU fused)
    int r = tid >> 4;                    // 0..15
    int c = (tid & 15) * 8;              // 0,8,...,120
    size_t rr = row0 + (size_t)r;
    if (rr >= (size_t)nrows) rr = (size_t)nrows - 1;   // clamp (no pad needed: 50000%16==0)
    const float* xp = x + rr * F + c;
    float4 a = ((const float4*)xp)[0];
    float4 b = ((const float4*)xp)[1];
    if (apply_relu) {
      a.x = fmaxf(a.x, 0.f); a.y = fmaxf(a.y, 0.f); a.z = fmaxf(a.z, 0.f); a.w = fmaxf(a.w, 0.f);
      b.x = fmaxf(b.x, 0.f); b.y = fmaxf(b.y, 0.f); b.z = fmaxf(b.z, 0.f); b.w = fmaxf(b.w, 0.f);
    }
    unsigned int* q = &ldsx[r * PAIRS + (c >> 1)];
    q[0] = pack2(a.x, a.y); q[1] = pack2(a.z, a.w);
    q[2] = pack2(b.x, b.y); q[3] = pack2(b.z, b.w);
  }
  __syncthreads();

  const int wave = tid >> 5;             // n-tile 0..7
  const int lane = tid & 31;
  const int m    = lane & 15;
  const int half = lane >> 4;

  const v8u* wp = (const v8u*)wpack;     // per-lane 32B B fragments
  v8f acc = {};
  #pragma unroll
  for (int s = 0; s < 4; ++s) {
    // A in this code's contiguous k order: VGPR j -> K=2j+16*half (matches B pack)
    v8u au;
    const unsigned int* ap = &ldsx[m * PAIRS + s * 16 + 8 * half];
    au[0] = ap[0];  au[1] = ap[1];  au[2] = ap[2];  au[3] = ap[3];
    au[4] = ap[4];  au[5] = ap[5];  au[6] = ap[6]; au[7] = ap[7];
    v8u bu = wp[(wave * 4 + s) * 32 + lane];
    acc = __builtin_amdgcn_wmma_f32_16x16x32_bf16(
        false, __builtin_bit_cast(v16bf, au),
        false, __builtin_bit_cast(v16bf, bu),
        (short)0, acc, false, false);
  }

  // D layout: VGPR r, lanes 0-15 -> M=r, lanes 16-31 -> M=r+8; N = lane&15
  size_t rowb = row0 + (size_t)(half * 8);
  if (rowb + 7 < (size_t)nrows) {
    float* hp = h + rowb * F + wave * 16 + m;
    #pragma unroll
    for (int r = 0; r < 8; ++r) hp[(size_t)r * F] = acc[r];
  }
}

// ---------------- agg = h * dis^2 + bias -----------------------------------
__global__ void agg_init_kernel(const float* __restrict__ h, const float* __restrict__ dis,
                                const float* __restrict__ bias, float* __restrict__ agg, int n) {
  int i = blockIdx.x * blockDim.x + threadIdx.x;       // over n*32 float4 slots
  if (i >= n * 32) return;
  int node = i >> 5, q = i & 31;
  float sn = dis[node]; sn *= sn;
  float4 hv = ((const float4*)h)[i];
  float4 bv = ((const float4*)bias)[q];
  float4 o;
  o.x = hv.x * sn + bv.x; o.y = hv.y * sn + bv.y;
  o.z = hv.z * sn + bv.z; o.w = hv.w * sn + bv.w;
  ((float4*)agg)[i] = o;
}

// ---------------- scatter: agg[dst] += h[src] * norm (wave per edge) -------
__global__ __launch_bounds__(256) void edge_scatter_kernel(
    const long long* __restrict__ edges, const float* __restrict__ dis,
    const float* __restrict__ h, float* __restrict__ agg, int E) {
  int wave = blockIdx.x * (blockDim.x >> 5) + (threadIdx.x >> 5);
  int lane = threadIdx.x & 31;
  if (wave >= E) return;
  int s = (int)edges[wave];
  int d = (int)edges[(size_t)E + wave];
  float norm = dis[s] * dis[d];
  float4 v = ((const float4*)(h + (size_t)s * F))[lane];
  float* a = agg + (size_t)d * F + lane * 4;
  atomic_add_f32(a + 0, v.x * norm);
  atomic_add_f32(a + 1, v.y * norm);
  atomic_add_f32(a + 2, v.z * norm);
  atomic_add_f32(a + 3, v.w * norm);
}

// ---------------- final 128->1 layer ---------------------------------------
__global__ __launch_bounds__(256) void final_dot_kernel(
    const float* __restrict__ x, const float* __restrict__ wout,
    float* __restrict__ h1, int n) {
  int node = blockIdx.x * 8 + (threadIdx.x >> 5);
  int lane = threadIdx.x & 31;
  if (node >= n) return;
  float4 xv = ((const float4*)(x + (size_t)node * F))[lane];
  xv.x = fmaxf(xv.x, 0.f); xv.y = fmaxf(xv.y, 0.f);
  xv.z = fmaxf(xv.z, 0.f); xv.w = fmaxf(xv.w, 0.f);
  float4 wv = ((const float4*)wout)[lane];
  float p = xv.x * wv.x + xv.y * wv.y + xv.z * wv.z + xv.w * wv.w;
  #pragma unroll
  for (int off = 16; off > 0; off >>= 1) p += __shfl_xor(p, off, 32);
  if (lane == 0) h1[node] = p;
}

__global__ void final_init_kernel(const float* __restrict__ h1, const float* __restrict__ dis,
                                  const float* __restrict__ bout, float* __restrict__ agg1, int n) {
  int i = blockIdx.x * blockDim.x + threadIdx.x;
  if (i >= n) return;
  float sn = dis[i]; sn *= sn;
  agg1[i] = h1[i] * sn + bout[0];
}

__global__ void final_scatter_kernel(const long long* __restrict__ edges,
                                     const float* __restrict__ dis,
                                     const float* __restrict__ h1,
                                     float* __restrict__ agg1, int E) {
  int e = blockIdx.x * blockDim.x + threadIdx.x;
  if (e >= E) return;
  int s = (int)edges[e];
  int d = (int)edges[(size_t)E + e];
  atomic_add_f32(&agg1[d], h1[s] * dis[s] * dis[d]);
}

__global__ void final_relu_kernel(const float* __restrict__ agg1, float* __restrict__ out, int n) {
  int i = blockIdx.x * blockDim.x + threadIdx.x;
  if (i < n) out[i] = fmaxf(agg1[i], 0.0f);
}

// ---------------------------------------------------------------------------
extern "C" void kernel_launch(void* const* d_in, const int* in_sizes, int n_in,
                              void* d_out, int out_size, void* d_ws, size_t ws_size,
                              hipStream_t stream) {
  const float*     features = (const float*)d_in[0];
  const float*     W        = (const float*)d_in[1];   // [3][128][128]
  const float*     b        = (const float*)d_in[2];   // [3][128]
  const float*     W_out    = (const float*)d_in[3];   // [128]
  const float*     b_out    = (const float*)d_in[4];   // [1]
  const long long* edges    = (const long long*)d_in[5]; // [2][E] int64

  const int N = in_sizes[0] / F;   // 50000
  const int E = in_sizes[5] / 2;   // 1600000
  (void)n_in; (void)out_size; (void)ws_size;

  char* ws = (char*)d_ws; size_t off = 0;
  auto carve = [&](size_t bytes) -> void* {
    off = (off + 255) & ~(size_t)255;
    void* p = ws + off; off += bytes; return p;
  };
  float*        dis   = (float*)carve((size_t)N * 4);
  float*        hbuf  = (float*)carve((size_t)N * F * 4);
  float*        buf0  = (float*)carve((size_t)N * F * 4);
  float*        buf1  = (float*)carve((size_t)N * F * 4);
  float*        h1    = (float*)carve((size_t)N * 4);
  float*        agg1  = (float*)carve((size_t)N * 4);
  unsigned int* wpack = (unsigned int*)carve((size_t)3 * 8192 * 4);

  dim3 blk(256);

  // 1) symmetric normalization: dis = (1 + in-degree)^-1/2
  zero_f32_kernel<<<(N + 255) / 256, blk, 0, stream>>>(dis, N);
  deg_kernel<<<(E + 255) / 256, blk, 0, stream>>>(edges, dis, E);
  dis_kernel<<<(N + 255) / 256, blk, 0, stream>>>(dis, N);

  // 2) pre-pack weights into WMMA bf16 B-fragment layout (amortized over 3125 blocks)
  pack_w_kernel<<<(3 * 8192) / 256, blk, 0, stream>>>(W, wpack);

  // 3) three GCN layers
  const int mblocks = (N + 15) / 16;       // 3125
  const int eblocks = (E + 7) / 8;         // 8 edge-waves per block
  const float* xs[3]  = {features, buf0, buf1};
  float*       aggs[3] = {buf0, buf1, buf0};
  for (int l = 0; l < 3; ++l) {
    gcn_gemm_bf16<<<mblocks, blk, 0, stream>>>(xs[l], wpack + (size_t)l * 8192, hbuf, l > 0, N);
    agg_init_kernel<<<(N * 32 + 255) / 256, blk, 0, stream>>>(hbuf, dis, b + (size_t)l * F, aggs[l], N);
    edge_scatter_kernel<<<eblocks, blk, 0, stream>>>(edges, dis, hbuf, aggs[l], E);
  }

  // 4) final 128->1 layer (x = relu(buf0))
  final_dot_kernel<<<(N + 7) / 8, blk, 0, stream>>>(buf0, W_out, h1, N);
  final_init_kernel<<<(N + 255) / 256, blk, 0, stream>>>(h1, dis, b_out, agg1, N);
  final_scatter_kernel<<<(E + 255) / 256, blk, 0, stream>>>(edges, dis, h1, agg1, E);
  final_relu_kernel<<<(N + 255) / 256, blk, 0, stream>>>(agg1, (float*)d_out, N);
}
